// KAN_EXTRACTOR_56384330662447
// MI455X (gfx1250) — matrix operand order repaired and checked by reading, not compiled, not run on hardware
//
#include <hip/hip_runtime.h>

typedef __attribute__((ext_vector_type(16))) __bf16 v16bf;
typedef __attribute__((ext_vector_type(8)))  __bf16 v8bf;
typedef __attribute__((ext_vector_type(8)))  float  v8f;

#define KAN_BATCH   16384
#define KAN_DOUT    64
#define KAN_MTILE   64
#define KAN_NTHR    128

// ---------------------------------------------------------------------------
// Fold ssp*coef (+ sb) into a transposed bf16 weight matrix WT[KAN_DOUT][9*din]:
//   WT[o][i]            = sb[i][o]                      (SiLU term)
//   WT[o][din + i*8+k]  = ssp[i][o] * coef[i][o][k]     (spline term)
// ---------------------------------------------------------------------------
__global__ void kan_wprep(const float* __restrict__ coef,
                          const float* __restrict__ sb,
                          const float* __restrict__ ssp,
                          __bf16* __restrict__ WT, int din) {
    const int kdim  = 9 * din;
    const int total = KAN_DOUT * kdim;
    int t = blockIdx.x * blockDim.x + threadIdx.x;
    if (t >= total) return;
    int o  = t / kdim;
    int kc = t - o * kdim;
    float v;
    if (kc < din) {
        v = sb[kc * KAN_DOUT + o];
    } else {
        int rem = kc - din;
        int i   = rem >> 3;
        int kk  = rem & 7;
        v = ssp[i * KAN_DOUT + o] * coef[(i * KAN_DOUT + o) * 8 + kk];
    }
    WT[o * kdim + kc] = (__bf16)v;
}

// ---------------------------------------------------------------------------
// One fused KAN layer: expand (SiLU + 8 cubic B-spline bases) -> LDS (bf16),
// then GEMM (64 x 9*DIN) x (9*DIN x 64) per block with v_wmma_f32_16x16x32_bf16.
// Wave ownership: each wave owns a 16-column N-stripe and all 4 M-tiles, so
// the global B-matrix (folded weights) is read once per block, and the 4x
// reuse happens out of LDS (A fragments, conflict-free padded stride).
// ---------------------------------------------------------------------------
template <int DIN>
__global__ __launch_bounds__(KAN_NTHR)
void kan_layer(const float* __restrict__ X,
               const float* __restrict__ grid,     // (DIN, 12) knot grid
               const __bf16* __restrict__ WT,      // (64, 9*DIN) transposed weights
               float* __restrict__ OUT) {          // (BATCH, 64)
    constexpr int KDIM   = 9 * DIN;
    constexpr int STRIDE = KDIM + 8;               // halves; 16B rows, bank-friendly

    extern __shared__ __align__(16) char smem_raw[];
    __bf16* F     = (__bf16*)smem_raw;
    float*  gridS = (float*)(smem_raw + (size_t)KAN_MTILE * STRIDE * sizeof(__bf16));

    const int tid  = threadIdx.x;
    const int row0 = blockIdx.x * KAN_MTILE;

    // stage knot grid in LDS (DIN x 12 fp32)
    for (int j = tid; j < DIN * 12; j += KAN_NTHR) gridS[j] = grid[j];
    __syncthreads();

    // ---- phase 1: feature expansion into LDS (bf16) ----
    for (int idx = tid; idx < KAN_MTILE * DIN; idx += KAN_NTHR) {
        int r = idx / DIN;
        int i = idx - r * DIN;
        float x = X[(size_t)(row0 + r) * DIN + i];

        float s = x / (1.0f + __expf(-x));         // SiLU

        float g[12];
        #pragma unroll
        for (int j = 0; j < 12; ++j) g[j] = gridS[i * 12 + j];

        float Bv[11];                               // Cox-de Boor, degree 0..3
        #pragma unroll
        for (int j = 0; j < 11; ++j)
            Bv[j] = (x >= g[j] && x < g[j + 1]) ? 1.0f : 0.0f;
        #pragma unroll
        for (int kk = 1; kk <= 3; ++kk) {
            #pragma unroll
            for (int j = 0; j < 11 - kk; ++j) {
                Bv[j] = (x - g[j])          / (g[j + kk]     - g[j])     * Bv[j]
                      + (g[j + kk + 1] - x) / (g[j + kk + 1] - g[j + 1]) * Bv[j + 1];
            }
        }

        __bf16* Frow = F + (size_t)r * STRIDE;
        Frow[i] = (__bf16)s;                        // SiLU block: cols [0, DIN)
        v8bf bb;
        #pragma unroll
        for (int k = 0; k < 8; ++k) bb[k] = (__bf16)Bv[k];
        *(v8bf*)(Frow + DIN + i * 8) = bb;          // spline block: 16B store
    }
    __syncthreads();

    // ---- phase 2: WMMA GEMM; wave = one 16-col N-stripe x all 64 rows ----
    const int wave = tid >> 5;
    const int lane = tid & 31;
    const int lr   = lane & 15;
    const int lh   = lane >> 4;
    const int n0   = wave * 16;

    v8f acc0 = {}, acc1 = {}, acc2 = {}, acc3 = {};

    // A layout (16-bit 16x32): lanes 0-15 -> K 0..7 / 16..23, lanes 16-31 -> K 8..15 / 24..31
    // Hoist one base pointer per M-tile so in-loop DS addressing is pure
    // immediate offsets (k0*2 <= 2304B fits the 16-bit DS offset field).
    const __bf16* a0 = F + (size_t)lr * STRIDE + lh * 8;
    const __bf16* a1 = a0 + (size_t)16 * STRIDE;
    const __bf16* a2 = a0 + (size_t)32 * STRIDE;
    const __bf16* a3 = a0 + (size_t)48 * STRIDE;
    // B layout (16-bit 32x16): lanes 0-15 -> K 0..7 / 16..23, lanes 16-31 -> K 8..15 / 24..31 for column n0+lr
    const __bf16* Wl = WT + (size_t)(n0 + lr) * KDIM + lh * 8;

    for (int k0 = 0; k0 < KDIM; k0 += 32) {
        union { v16bf v; v8bf h[2]; } b;
        b.h[0] = *(const v8bf*)(Wl + k0);
        b.h[1] = *(const v8bf*)(Wl + k0 + 16);
        #define KAN_DO_TILE(ACC, AP) {                                          \
            union { v16bf v; v8bf h[2]; } a;                                    \
            a.h[0] = *(const v8bf*)((AP) + k0);                                 \
            a.h[1] = *(const v8bf*)((AP) + k0 + 16);                            \
            ACC = __builtin_amdgcn_wmma_f32_16x16x32_bf16(                      \
                false, a.v, false, b.v, (short)0, ACC, false, false); }
        KAN_DO_TILE(acc0, a0)
        KAN_DO_TILE(acc1, a1)
        KAN_DO_TILE(acc2, a2)
        KAN_DO_TILE(acc3, a3)
        #undef KAN_DO_TILE
    }

    // ---- phase 3: store D (f32 16x16: M = vgpr + 8*lh, N = lr) ----
    float* outBase = OUT + (size_t)row0 * KAN_DOUT + n0 + lr;
    #pragma unroll
    for (int j = 0; j < 8; ++j) {
        int m = j + 8 * lh;
        outBase[(size_t)(m)      * KAN_DOUT] = acc0[j];
        outBase[(size_t)(m + 16) * KAN_DOUT] = acc1[j];
        outBase[(size_t)(m + 32) * KAN_DOUT] = acc2[j];
        outBase[(size_t)(m + 48) * KAN_DOUT] = acc3[j];
    }
}

// ---------------------------------------------------------------------------
// d_in layout (setup_inputs order):
//   [0] features | [1..4] grids_pi | [5..8] coefs_pi | [9..12] sbs_pi
//   [13..16] ssps_pi | [17..20] grids_vf | [21..24] coefs_vf
//   [25..28] sbs_vf | [29..32] ssps_vf
// ---------------------------------------------------------------------------
extern "C" void kernel_launch(void* const* d_in, const int* in_sizes, int n_in,
                              void* d_out, int out_size, void* d_ws, size_t ws_size,
                              hipStream_t stream) {
    (void)in_sizes; (void)n_in; (void)out_size; (void)ws_size;
    const float* features = (const float*)d_in[0];
    const int dins[4] = {128, 64, 64, 64};

    char* ws = (char*)d_ws;

    // workspace: 8 folded-weight matrices, then two ping-pong activation buffers
    size_t wt_off[8]; size_t off = 0;
    for (int s = 0; s < 8; ++s) {
        wt_off[s] = off;
        off += (size_t)KAN_DOUT * 9 * dins[s & 3] * sizeof(__bf16);
        off  = (off + 255) & ~(size_t)255;
    }
    float* actA = (float*)(ws + off);
    float* actB = actA + (size_t)KAN_BATCH * KAN_DOUT;

    for (int net = 0; net < 2; ++net) {
        const int gbase = 1 + net * 16;
        const int cbase = gbase + 4;
        const int sbase = cbase + 4;
        const int pbase = sbase + 4;

        // fold weights for all 4 layers of this net
        for (int l = 0; l < 4; ++l) {
            const int total = KAN_DOUT * 9 * dins[l];
            __bf16* WT = (__bf16*)(ws + wt_off[net * 4 + l]);
            kan_wprep<<<(total + 255) / 256, 256, 0, stream>>>(
                (const float*)d_in[cbase + l], (const float*)d_in[sbase + l],
                (const float*)d_in[pbase + l], WT, dins[l]);
        }

        // run the 4 chained layers
        float* out_final = (float*)d_out + (size_t)net * KAN_BATCH * KAN_DOUT;
        const float* x = features;
        for (int l = 0; l < 4; ++l) {
            float* y = (l == 3) ? out_final : ((l & 1) ? actB : actA);
            __bf16* WT    = (__bf16*)(ws + wt_off[net * 4 + l]);
            const float* g = (const float*)d_in[gbase + l];
            const size_t lds = (size_t)KAN_MTILE * (9 * dins[l] + 8) * sizeof(__bf16)
                             + (size_t)dins[l] * 12 * sizeof(float);
            if (dins[l] == 128)
                kan_layer<128><<<KAN_BATCH / KAN_MTILE, KAN_NTHR, lds, stream>>>(x, g, WT, y);
            else
                kan_layer<64><<<KAN_BATCH / KAN_MTILE, KAN_NTHR, lds, stream>>>(x, g, WT, y);
            x = y;
        }
    }
}
